// Qwen2ManualForward_37950331028130
// MI455X (gfx1250) — matrix operand order repaired and checked by reading, not compiled, not run on hardware
//
#include <hip/hip_runtime.h>

// ---------------------------------------------------------------------------
// Qwen2-style 2-layer decoder forward for gfx1250 (MI455X).
// Matmuls via v_wmma_f32_16x16x32_f16 (fp16 in, fp32 accumulate).
// Attention K/V tiles staged with GLOBAL_LOAD_ASYNC_TO_LDS_B128 from an
// fp16 shadow KV cache (V pre-transposed), synchronized via ASYNCcnt.
// ---------------------------------------------------------------------------

#define CL 2
#define CH 2048
#define CNH 16
#define CNKV 2
#define CHD 128
#define CS 1024
#define CP 1024
#define CT 2048
#define CFF 5632
#define CG 8
#define CSCALE 0.08838834764831845f   /* 1/sqrt(128) */
#define CLOGTHETA 13.815510557964274f /* ln(1e6) */

typedef __attribute__((ext_vector_type(16))) _Float16 v16h;
typedef __attribute__((ext_vector_type(8)))  float    v8f;

union FragH {
    v16h v;
    _Float16 e[16];
    uint4 u[2];
};

__device__ __forceinline__ v8f wmma16(v16h a, v16h b, v8f c) {
    return __builtin_amdgcn_wmma_f32_16x16x32_f16(
        false, a, false, b, (short)0, c, false, false);
}

__device__ __forceinline__ v8f vzero8() {
    v8f z = {0.f, 0.f, 0.f, 0.f, 0.f, 0.f, 0.f, 0.f};
    return z;
}

// 16-byte async copy global -> LDS (CDNA5 ASYNCcnt path). Each lane supplies
// its own LDS byte offset (low 32 bits of the generic shared pointer) and
// 64-bit global address.
__device__ __forceinline__ void async_copy16(void* lds_dst, const void* gsrc) {
    unsigned lds_off = (unsigned)(size_t)lds_dst;
    asm volatile("global_load_async_to_lds_b128 %0, %1, off"
                 :: "v"(lds_off), "v"(gsrc)
                 : "memory");
}

__device__ __forceinline__ void async_wait0() {
    asm volatile("s_wait_asynccnt 0x0" ::: "memory");
}

// ---------------------------------------------------------------------------
// Tiled GEMM: C[M,N] = A[M,K] @ B[K,N] (+bias) (+C)
// 128x128x32 tiles, 256 threads (8 waves), each wave a 32x64 patch.
// flags bit0: add bias[col]; bit1: accumulate into existing C.
// ---------------------------------------------------------------------------
__global__ __launch_bounds__(256) void gemm_kernel(
    const float* __restrict__ A, int lda,
    const float* __restrict__ B, int ldb,
    const float* __restrict__ bias,
    float* __restrict__ C, int ldc,
    int K, int flags)
{
    __shared__ _Float16 As[128][40];   // [m][k], padded rows (80B, 16B-aligned)
    __shared__ _Float16 Bs[128][40];   // [n][k] transposed

    const int tid  = threadIdx.x;
    const int lane = tid & 31;
    const int wid  = tid >> 5;
    const int g    = lane >> 4;
    const int ln   = lane & 15;
    const int wm   = wid >> 1;        // 0..3
    const int wn   = wid & 1;         // 0..1
    const int m0   = blockIdx.y * 128;
    const int n0   = blockIdx.x * 128;

    v8f acc[2][4];
#pragma unroll
    for (int i = 0; i < 2; ++i)
#pragma unroll
        for (int j = 0; j < 4; ++j) acc[i][j] = vzero8();

    for (int k0 = 0; k0 < K; k0 += 32) {
        __syncthreads();
#pragma unroll
        for (int i = 0; i < 16; ++i) {          // A tile 128x32
            int e = tid + (i << 8);
            int r = e >> 5, c = e & 31;
            As[r][c] = (_Float16)A[(size_t)(m0 + r) * lda + (k0 + c)];
        }
#pragma unroll
        for (int i = 0; i < 16; ++i) {          // B tile 32x128 -> transposed
            int e = tid + (i << 8);
            int r = e >> 7, c = e & 127;
            Bs[c][r] = (_Float16)B[(size_t)(k0 + r) * ldb + (n0 + c)];
        }
        __syncthreads();

        FragH a[2], b[4];
#pragma unroll
        for (int i = 0; i < 2; ++i) {
            int m = wm * 32 + i * 16 + ln;
            a[i].u[0] = *reinterpret_cast<const uint4*>(&As[m][g * 8]);
            a[i].u[1] = *reinterpret_cast<const uint4*>(&As[m][g * 8 + 16]);
        }
#pragma unroll
        for (int j = 0; j < 4; ++j) {
            int n = wn * 64 + j * 16 + ln;
            b[j].u[0] = *reinterpret_cast<const uint4*>(&Bs[n][g * 8]);
            b[j].u[1] = *reinterpret_cast<const uint4*>(&Bs[n][g * 8 + 16]);
        }
#pragma unroll
        for (int i = 0; i < 2; ++i)
#pragma unroll
            for (int j = 0; j < 4; ++j)
                acc[i][j] = wmma16(a[i].v, b[j].v, acc[i][j]);
    }

#pragma unroll
    for (int i = 0; i < 2; ++i) {
#pragma unroll
        for (int j = 0; j < 4; ++j) {
            int col  = n0 + wn * 64 + j * 16 + ln;
            int rowb = m0 + wm * 32 + i * 16 + 8 * g;
            float bv = (flags & 1) ? bias[col] : 0.0f;
#pragma unroll
            for (int r = 0; r < 8; ++r) {
                size_t idx = (size_t)(rowb + r) * ldc + col;
                float v = acc[i][j][r] + bv;
                if (flags & 2) v += C[idx];
                C[idx] = v;
            }
        }
    }
}

// ---------------------------------------------------------------------------
// Flash-style GQA attention. grid.x = NKV * (S/16). Block = 8 waves = the 8
// query heads of one kv head, 16 query rows. K/V tiles async-copied from the
// fp16 shadow KV cache into LDS and shared by all 8 waves.
// ---------------------------------------------------------------------------
__global__ __launch_bounds__(256) void attn_kernel(
    const float* __restrict__ q,          // [S, NH*HD], roped, fp32
    const _Float16* __restrict__ k16,     // [NKV, T, HD] fp16 keys (roped)
    const _Float16* __restrict__ v16t,    // [NKV, HD, T] fp16 values, transposed
    float* __restrict__ out)              // [S, NH*HD]
{
    __shared__ _Float16 Kt[32][136];      // [key][dim], padded rows
    __shared__ _Float16 Vt[128][40];      // [dim][key], padded rows
    __shared__ _Float16 Pb[8][16][40];    // per-wave probs [m][key]

    const int tid   = threadIdx.x;
    const int lane  = tid & 31;
    const int wid   = tid >> 5;
    const int g     = lane >> 4;
    const int ln    = lane & 15;
    const int kv    = blockIdx.x >> 6;           // / (S/16)
    const int qbase = (blockIdx.x & 63) * 16;
    const int h     = kv * CG + wid;
    const _Float16* khead  = k16  + (size_t)kv * CT * CHD;
    const _Float16* vheadT = v16t + (size_t)kv * CHD * CT;

    // Preload Q fragments (16 rows x 128 dims, 4 k-steps of 32)
    FragH qf[4];
#pragma unroll
    for (int t = 0; t < 4; ++t) {
        const float* qp = q + (size_t)(qbase + ln) * (CNH * CHD) + h * CHD + t * 32 + g * 8;
#pragma unroll
        for (int ii = 0; ii < 8; ++ii) qf[t].e[ii]     = (_Float16)qp[ii];
#pragma unroll
        for (int ii = 0; ii < 8; ++ii) qf[t].e[8 + ii] = (_Float16)qp[16 + ii];
    }

    v8f o[8];
    float mrow[8], lrow[8];
#pragma unroll
    for (int c = 0; c < 8; ++c) o[c] = vzero8();
#pragma unroll
    for (int r = 0; r < 8; ++r) { mrow[r] = -3.0e38f; lrow[r] = 0.0f; }

    const int qmax = CP + qbase + 15;
    const int nkt  = (qmax >> 5) + 1;   // key tiles of 32 with any unmasked key

    for (int kt = 0; kt < nkt; ++kt) {
        const int t0 = kt * 32;
        __syncthreads();
        // ---- async stage K tile: 32x128 fp16 = 512 x 16B chunks ----
#pragma unroll
        for (int i = 0; i < 2; ++i) {
            int c = tid + (i << 8);
            int r = c >> 4, qq = c & 15;
            async_copy16(&Kt[r][qq * 8],
                         khead + (size_t)(t0 + r) * CHD + qq * 8);
        }
        // ---- async stage V tile transposed: 128 dims x 32 keys ----
#pragma unroll
        for (int i = 0; i < 2; ++i) {
            int c = tid + (i << 8);
            int dim = c >> 2, qq = c & 3;
            async_copy16(&Vt[dim][qq * 8],
                         vheadT + (size_t)dim * CT + t0 + qq * 8);
        }
        async_wait0();
        __syncthreads();

        // scores: 16 queries x 32 keys
        v8f s0 = vzero8(), s1 = vzero8();
#pragma unroll
        for (int t = 0; t < 4; ++t) {
            int klo = t * 32 + g * 8;
            FragH b0, b1;
            b0.u[0] = *reinterpret_cast<const uint4*>(&Kt[ln][klo]);
            b0.u[1] = *reinterpret_cast<const uint4*>(&Kt[ln][klo + 16]);
            b1.u[0] = *reinterpret_cast<const uint4*>(&Kt[16 + ln][klo]);
            b1.u[1] = *reinterpret_cast<const uint4*>(&Kt[16 + ln][klo + 16]);
            s0 = wmma16(qf[t].v, b0.v, s0);
            s1 = wmma16(qf[t].v, b1.v, s1);
        }

        // online softmax per query row (row == r + 8*g, key == lane%16 (+16))
        float p0a[8], p1a[8], alpha[8];
#pragma unroll
        for (int r = 0; r < 8; ++r) {
            int m = r + 8 * g;
            int qpos = CP + qbase + m;
            float x0 = s0[r] * CSCALE + (((t0 + ln)      > qpos) ? -65504.0f : 0.0f);
            float x1 = s1[r] * CSCALE + (((t0 + 16 + ln) > qpos) ? -65504.0f : 0.0f);
            float mt = fmaxf(x0, x1);
            mt = fmaxf(mt, __shfl_xor(mt, 1));
            mt = fmaxf(mt, __shfl_xor(mt, 2));
            mt = fmaxf(mt, __shfl_xor(mt, 4));
            mt = fmaxf(mt, __shfl_xor(mt, 8));
            float mnew = fmaxf(mrow[r], mt);
            float al = __expf(mrow[r] - mnew);
            float p0 = __expf(x0 - mnew);
            float p1 = __expf(x1 - mnew);
            float rs = p0 + p1;
            rs += __shfl_xor(rs, 1);
            rs += __shfl_xor(rs, 2);
            rs += __shfl_xor(rs, 4);
            rs += __shfl_xor(rs, 8);
            lrow[r] = lrow[r] * al + rs;
            mrow[r] = mnew;
            alpha[r] = al; p0a[r] = p0; p1a[r] = p1;
        }
#pragma unroll
        for (int c = 0; c < 8; ++c)
#pragma unroll
            for (int r = 0; r < 8; ++r) o[c][r] *= alpha[r];

        // route probs through LDS into A-fragment layout
#pragma unroll
        for (int r = 0; r < 8; ++r) {
            Pb[wid][r + 8 * g][ln]      = (_Float16)p0a[r];
            Pb[wid][r + 8 * g][16 + ln] = (_Float16)p1a[r];
        }
        __syncthreads();

        FragH pf;
        pf.u[0] = *reinterpret_cast<const uint4*>(&Pb[wid][ln][g * 8]);
        pf.u[1] = *reinterpret_cast<const uint4*>(&Pb[wid][ln][g * 8 + 16]);
#pragma unroll
        for (int c = 0; c < 8; ++c) {    // O(16x128) += P(16x32) @ V(32x128)
            FragH vf;
            int dim = c * 16 + ln;
            vf.u[0] = *reinterpret_cast<const uint4*>(&Vt[dim][g * 8]);
            vf.u[1] = *reinterpret_cast<const uint4*>(&Vt[dim][g * 8 + 16]);
            o[c] = wmma16(pf.v, vf.v, o[c]);
        }
    }

#pragma unroll
    for (int c = 0; c < 8; ++c) {
        int col = h * CHD + c * 16 + ln;
#pragma unroll
        for (int r = 0; r < 8; ++r) {
            int row = qbase + r + 8 * g;
            out[(size_t)row * (CNH * CHD) + col] = o[c][r] / lrow[r];
        }
    }
}

// ---------------------------------------------------------------------------
// RMSNorm: one block per row of H=2048.
// ---------------------------------------------------------------------------
__global__ __launch_bounds__(256) void rmsnorm_kernel(
    const float* __restrict__ in, const float* __restrict__ w,
    float* __restrict__ out)
{
    __shared__ float red[256];
    const int row = blockIdx.x, tid = threadIdx.x;
    const float* x = in + (size_t)row * CH;
    float s = 0.0f;
    for (int c = tid; c < CH; c += 256) { float v = x[c]; s += v * v; }
    red[tid] = s;
    __syncthreads();
    for (int st = 128; st > 0; st >>= 1) {
        if (tid < st) red[tid] += red[tid + st];
        __syncthreads();
    }
    float r = rsqrtf(red[0] / (float)CH + 1e-6f);
    float* y = out + (size_t)row * CH;
    for (int c = tid; c < CH; c += 256) y[c] = x[c] * r * w[c];
}

// ---------------------------------------------------------------------------
// RoPE on Q in place: id over S*NH*64
// ---------------------------------------------------------------------------
__global__ void rope_q_kernel(float* __restrict__ q, const int* __restrict__ pid)
{
    int id = blockIdx.x * 256 + threadIdx.x;
    int j = id & 63;
    int t = id >> 6;
    int h = t & (CNH - 1);
    int s = t >> 4;
    float pos = (float)pid[s];
    float freq = pos * __expf(-(float)(2 * j) * (1.0f / (float)CHD) * CLOGTHETA);
    float sn, cs;
    __sincosf(freq, &sn, &cs);
    size_t base = (size_t)s * (CNH * CHD) + (size_t)h * CHD;
    float x1 = q[base + j], x2 = q[base + 64 + j];
    q[base + j]      = x1 * cs - x2 * sn;
    q[base + 64 + j] = x2 * cs + x1 * sn;
}

// RoPE on new K rows -> fp32 present output AND fp16 shadow cache.
__global__ void rope_k_store_kernel(const float* __restrict__ ktmp,
                                    const int* __restrict__ pid,
                                    float* __restrict__ dstk,
                                    _Float16* __restrict__ k16)
{
    int id = blockIdx.x * 256 + threadIdx.x;   // S*NKV*64
    int j = id & 63;
    int t = id >> 6;
    int kvh = t & (CNKV - 1);
    int s = t >> 1;
    float pos = (float)pid[s];
    float freq = pos * __expf(-(float)(2 * j) * (1.0f / (float)CHD) * CLOGTHETA);
    float sn, cs;
    __sincosf(freq, &sn, &cs);
    const float* src = ktmp + (size_t)s * (CNKV * CHD) + (size_t)kvh * CHD;
    float x1 = src[j], x2 = src[64 + j];
    float r0 = x1 * cs - x2 * sn;
    float r1 = x2 * cs + x1 * sn;
    size_t rowoff = (size_t)kvh * CT * CHD + (size_t)(CP + s) * CHD;
    dstk[rowoff + j]      = r0;
    dstk[rowoff + 64 + j] = r1;
    k16[rowoff + j]       = (_Float16)r0;
    k16[rowoff + 64 + j]  = (_Float16)r1;
}

// New V rows -> fp32 present output AND fp16 transposed shadow cache.
__global__ void v_store_kernel(const float* __restrict__ vtmp,
                               float* __restrict__ dstv,
                               _Float16* __restrict__ v16t)
{
    int id = blockIdx.x * 256 + threadIdx.x;   // S*NKV*HD
    int d = id & 127;
    int t = id >> 7;
    int kvh = t & (CNKV - 1);
    int s = t >> 1;
    float val = vtmp[(size_t)s * (CNKV * CHD) + (size_t)kvh * CHD + d];
    dstv[(size_t)kvh * CT * CHD + (size_t)(CP + s) * CHD + d] = val;
    v16t[((size_t)kvh * CHD + d) * CT + (CP + s)] = (_Float16)val;
}

// Past K -> fp32 present output AND fp16 shadow (same row-major layout).
__global__ void copy_past_k_kernel(const float* __restrict__ src,
                                   float* __restrict__ dst,
                                   _Float16* __restrict__ k16)
{
    int id = blockIdx.x * 256 + threadIdx.x;   // NKV*P*HD = 262144
    int kvh = id >> 17;                        // P*HD = 131072
    int rem = id & 131071;
    float v = src[id];
    size_t o = (size_t)kvh * CT * CHD + rem;
    dst[o] = v;
    k16[o] = (_Float16)v;
}

// Past V -> fp32 present output AND fp16 transposed shadow.
__global__ void copy_past_v_kernel(const float* __restrict__ src,
                                   float* __restrict__ dst,
                                   _Float16* __restrict__ v16t)
{
    int id = blockIdx.x * 256 + threadIdx.x;   // NKV*P*HD = 262144
    int kvh = id >> 17;
    int rem = id & 131071;
    int t = rem >> 7;
    int d = rem & 127;
    float v = src[id];
    dst[(size_t)kvh * CT * CHD + rem] = v;
    v16t[((size_t)kvh * CHD + d) * CT + t] = (_Float16)v;
}

__global__ void silu_mul_kernel(float* __restrict__ gate, const float* __restrict__ up)
{
    int id = blockIdx.x * 256 + threadIdx.x;   // S*FF exactly
    float g = gate[id];
    gate[id] = g / (1.0f + __expf(-g)) * up[id];
}

// ---------------------------------------------------------------------------
extern "C" void kernel_launch(void* const* d_in, const int* in_sizes, int n_in,
                              void* d_out, int out_size, void* d_ws, size_t ws_size,
                              hipStream_t stream)
{
    (void)in_sizes; (void)n_in; (void)out_size; (void)ws_size;

    const float* embeds = (const float*)d_in[0];
    const int*   pid    = (const int*)d_in[1];
    const float* past_k = (const float*)d_in[2];
    const float* past_v = (const float*)d_in[3];
    const float* ln1    = (const float*)d_in[4];
    const float* wq     = (const float*)d_in[5];
    const float* bq     = (const float*)d_in[6];
    const float* wk     = (const float*)d_in[7];
    const float* bk     = (const float*)d_in[8];
    const float* wv     = (const float*)d_in[9];
    const float* bv     = (const float*)d_in[10];
    const float* wo     = (const float*)d_in[11];
    const float* ln2    = (const float*)d_in[12];
    const float* wg     = (const float*)d_in[13];
    const float* wu     = (const float*)d_in[14];
    const float* wd     = (const float*)d_in[15];
    const float* normw  = (const float*)d_in[16];
    float* out = (float*)d_out;

    const size_t SH   = (size_t)CS * CH;         // 2,097,152
    const size_t KVSZ = (size_t)CNKV * CT * CHD; // 524,288

    float* ws = (float*)d_ws;
    size_t off = 0;
    float* hid  = ws + off; off += SH;
    float* X    = ws + off; off += SH;
    float* qb   = ws + off; off += SH;                  // [S, NH*HD]
    float* ktmp = ws + off; off += (size_t)CS * CNKV * CHD;
    float* vtmp = ws + off; off += (size_t)CS * CNKV * CHD;
    float* att  = ws + off; off += SH;
    float* gate = ws + off; off += (size_t)CS * CFF;
    float* up   = ws + off; off += (size_t)CS * CFF;
    _Float16* k16  = (_Float16*)(ws + off); off += KVSZ / 2;  // fp16 shadow K
    _Float16* v16t = (_Float16*)(ws + off); off += KVSZ / 2;  // fp16 shadow V (transposed)

    hipMemcpyAsync(hid, embeds, SH * sizeof(float), hipMemcpyDeviceToDevice, stream);

    for (int i = 0; i < CL; ++i) {
        float* outk = out + SH + (size_t)i * 2 * KVSZ;
        float* outv = outk + KVSZ;

        rmsnorm_kernel<<<CS, 256, 0, stream>>>(hid, ln1 + (size_t)i * CH, X);

        gemm_kernel<<<dim3(16, 8), 256, 0, stream>>>(
            X, CH, wq + (size_t)i * CH * (CNH * CHD), CNH * CHD,
            bq + (size_t)i * (CNH * CHD), qb, CNH * CHD, CH, 1);
        gemm_kernel<<<dim3(2, 8), 256, 0, stream>>>(
            X, CH, wk + (size_t)i * CH * (CNKV * CHD), CNKV * CHD,
            bk + (size_t)i * (CNKV * CHD), ktmp, CNKV * CHD, CH, 1);
        gemm_kernel<<<dim3(2, 8), 256, 0, stream>>>(
            X, CH, wv + (size_t)i * CH * (CNKV * CHD), CNKV * CHD,
            bv + (size_t)i * (CNKV * CHD), vtmp, CNKV * CHD, CH, 1);

        rope_q_kernel<<<(CS * CNH * 64) / 256, 256, 0, stream>>>(qb, pid);
        copy_past_k_kernel<<<(CNKV * CP * CHD) / 256, 256, 0, stream>>>(
            past_k + (size_t)i * CNKV * CP * CHD, outk, k16);
        copy_past_v_kernel<<<(CNKV * CP * CHD) / 256, 256, 0, stream>>>(
            past_v + (size_t)i * CNKV * CP * CHD, outv, v16t);
        rope_k_store_kernel<<<(CS * CNKV * 64) / 256, 256, 0, stream>>>(ktmp, pid, outk, k16);
        v_store_kernel<<<(CS * CNKV * CHD) / 256, 256, 0, stream>>>(vtmp, outv, v16t);

        attn_kernel<<<CNKV * (CS / 16), 256, 0, stream>>>(qb, k16, v16t, att);

        gemm_kernel<<<dim3(16, 8), 256, 0, stream>>>(
            att, CNH * CHD, wo + (size_t)i * (CNH * CHD) * CH, CH,
            nullptr, hid, CH, CNH * CHD, 2);

        rmsnorm_kernel<<<CS, 256, 0, stream>>>(hid, ln2 + (size_t)i * CH, X);

        gemm_kernel<<<dim3(44, 8), 256, 0, stream>>>(
            X, CH, wg + (size_t)i * CH * CFF, CFF, nullptr, gate, CFF, CH, 0);
        gemm_kernel<<<dim3(44, 8), 256, 0, stream>>>(
            X, CH, wu + (size_t)i * CH * CFF, CFF, nullptr, up, CFF, CH, 0);
        silu_mul_kernel<<<((size_t)CS * CFF) / 256, 256, 0, stream>>>(gate, up);
        gemm_kernel<<<dim3(16, 8), 256, 0, stream>>>(
            gate, CFF, wd + (size_t)i * CFF * CH, CH, nullptr, hid, CH, CFF, 2);
    }

    rmsnorm_kernel<<<CS, 256, 0, stream>>>(hid, normw, out);
}
